// ContextQueryAttention_54649163874968
// MI455X (gfx1250) — compile-verified, matrix operand order checked by reading, not run on hardware
//
#include <hip/hip_runtime.h>
#include <math.h>

typedef __attribute__((ext_vector_type(16))) __bf16 v16bf;
typedef __attribute__((ext_vector_type(8)))  float  v8f;

static constexpr int B_ = 32, D_ = 128, C_ = 1024, Q_ = 128;
static constexpr int PITCH = 36;   // LDS row pitch (floats): 144B, 16B-aligned, odd/4 -> conflict-free

// K index for slot e (0..15) of the 16-bit WMMA A/B operand layout (ISA 7.12.2):
// slots 0..7 -> K = hi*8 + e (contiguous 8), slots 8..15 -> K = 16 + hi*8 + (e-8).
__device__ __forceinline__ int kslot(int e, int hi) {
  return ((e < 8) ? 0 : 16) + hi * 8 + (e & 7);
}

// pack two float4 (8 consecutive K values) into slots [base, base+8) of a bf16 fragment
__device__ __forceinline__ void pack8(v16bf& v, int base, float4 x, float4 y) {
  v[base + 0] = (__bf16)x.x; v[base + 1] = (__bf16)x.y;
  v[base + 2] = (__bf16)x.z; v[base + 3] = (__bf16)x.w;
  v[base + 4] = (__bf16)y.x; v[base + 5] = (__bf16)y.y;
  v[base + 6] = (__bf16)y.z; v[base + 7] = (__bf16)y.w;
}

// gfx1250 async global->LDS copy (ASYNCcnt), ISA 15.18.3 op 97
__device__ __forceinline__ void async_copy_b64(unsigned lds_byte_off, const float* g) {
  asm volatile("global_load_async_to_lds_b64 %0, %1, off" :: "v"(lds_byte_off), "v"(g) : "memory");
}
__device__ __forceinline__ void wait_async0() {
  asm volatile("s_wait_asynccnt 0x0" ::: "memory");
}

// ---------------- part_c[b,i] = sum_d w_c[d] * ctx[b,d,i] ----------------
__global__ void partc_kernel(const float* __restrict__ ctx,
                             const float* __restrict__ w,
                             float* __restrict__ out) {
  int idx = blockIdx.x * blockDim.x + threadIdx.x;
  if (idx >= B_ * C_) return;
  int b = idx / C_, i = idx % C_;
  float acc = 0.f;
#pragma unroll 4
  for (int d = 0; d < D_; ++d) acc += w[d] * ctx[(b * D_ + d) * C_ + i];
  out[idx] = acc;
}

// ---------------- part_q[b,j] = sum_d w_q[d] * q[b,d,j] ----------------
__global__ void partq_kernel(const float* __restrict__ q,
                             const float* __restrict__ w,
                             float* __restrict__ out) {
  int idx = blockIdx.x * blockDim.x + threadIdx.x;
  if (idx >= B_ * Q_) return;
  int b = idx / Q_, j = idx % Q_;
  float acc = 0.f;
#pragma unroll 4
  for (int d = 0; d < D_; ++d) acc += w[d] * q[(b * D_ + d) * Q_ + j];
  out[idx] = acc;
}

// ------- S[b,i,j] = ctx^T diag(w_cq) q + pc + pq + bias --------------------
// Block: 8 waves, macro-tile 128(i) x 16(j), K=D staged through LDS transposed.
__global__ void s_gemm_kernel(const float* __restrict__ ctx,
                              const float* __restrict__ q,
                              const float* __restrict__ w_cq,
                              const float* __restrict__ pc,
                              const float* __restrict__ pq,
                              const float* __restrict__ bias,
                              float* __restrict__ S) {
  __shared__ float lds_ctx[128 * PITCH];   // [i_local][d] 18.4 KB
  __shared__ float lds_q[16 * PITCH];      // [j_local][d] (w_cq folded)

  int tid  = threadIdx.x;
  int w    = tid >> 5;
  int lane = tid & 31;
  int hi = lane >> 4, ln = lane & 15;

  int b   = blockIdx.x >> 6;               // 64 macro-tiles per batch
  int rem = blockIdx.x & 63;
  int im  = (rem >> 3) * 128;              // i macro base
  int j0  = (rem & 7) * 16;

  v8f acc = {};
#pragma unroll
  for (int k0 = 0; k0 < D_; k0 += 32) {
    __syncthreads();
    // stage ctx slab 32(d) x 128(i), coalesced b128 in, transposed out
#pragma unroll
    for (int s = 0; s < 4; ++s) {
      int flat = s * 1024 + tid * 4;       // 0..4095
      int d = flat >> 7, i = flat & 127;
      float4 v = *(const float4*)&ctx[(b * D_ + k0 + d) * C_ + im + i];
      lds_ctx[(i + 0) * PITCH + d] = v.x;
      lds_ctx[(i + 1) * PITCH + d] = v.y;
      lds_ctx[(i + 2) * PITCH + d] = v.z;
      lds_ctx[(i + 3) * PITCH + d] = v.w;
      if (k0 + 32 < D_)                    // prefetch next K slab (global_prefetch)
        __builtin_prefetch(&ctx[(b * D_ + k0 + 32 + d) * C_ + im + i], 0, 1);
    }
    // stage question slab 32(d) x 16(j), fold w_cq
    if (tid < 128) {
      int flat = tid * 4;
      int d = flat >> 4, jj = flat & 15;
      float4 v = *(const float4*)&q[(b * D_ + k0 + d) * Q_ + j0 + jj];
      float wv = w_cq[k0 + d];
      lds_q[(jj + 0) * PITCH + d] = v.x * wv;
      lds_q[(jj + 1) * PITCH + d] = v.y * wv;
      lds_q[(jj + 2) * PITCH + d] = v.z * wv;
      lds_q[(jj + 3) * PITCH + d] = v.w * wv;
    }
    __syncthreads();

    const float* ar = &lds_ctx[(w * 16 + ln) * PITCH];  // A row m = i
    const float* br = &lds_q[ln * PITCH];               // B col n = j
    float4 a0 = *(const float4*)&ar[hi * 8];
    float4 a1 = *(const float4*)&ar[hi * 8 + 4];
    float4 a2 = *(const float4*)&ar[16 + hi * 8];
    float4 a3 = *(const float4*)&ar[20 + hi * 8];
    float4 b0 = *(const float4*)&br[hi * 8];
    float4 b1 = *(const float4*)&br[hi * 8 + 4];
    float4 b2 = *(const float4*)&br[16 + hi * 8];
    float4 b3 = *(const float4*)&br[20 + hi * 8];
    v16bf A, Bv;
    pack8(A, 0, a0, a1);  pack8(A, 8, a2, a3);
    pack8(Bv, 0, b0, b1); pack8(Bv, 8, b2, b3);
    acc = __builtin_amdgcn_wmma_f32_16x16x32_bf16(false, A, false, Bv,
                                                  (short)0, acc, false, false);
  }

  int i0 = im + w * 16;
  float pqv = pq[b * Q_ + j0 + ln] + bias[0];
#pragma unroll
  for (int r = 0; r < 8; ++r) {
    int m = r + hi * 8;
    S[(b * C_ + i0 + m) * Q_ + j0 + ln] = acc[r] + pc[b * C_ + i0 + m] + pqv;
  }
}

// ------------- row softmax stats over j (Q=128): one wave per row -------------
__global__ void rowstats_kernel(const float* __restrict__ S,
                                float* __restrict__ rmax,
                                float* __restrict__ rrcp) {
  int gw   = (blockIdx.x * blockDim.x + threadIdx.x) >> 5;
  int lane = threadIdx.x & 31;
  if (gw >= B_ * C_) return;
  const float* row = S + (size_t)gw * Q_;
  float v0 = row[lane], v1 = row[lane + 32], v2 = row[lane + 64], v3 = row[lane + 96];
  float m = fmaxf(fmaxf(v0, v1), fmaxf(v2, v3));
#pragma unroll
  for (int off = 16; off > 0; off >>= 1) m = fmaxf(m, __shfl_xor(m, off, 32));
  float s = __expf(v0 - m) + __expf(v1 - m) + __expf(v2 - m) + __expf(v3 - m);
#pragma unroll
  for (int off = 16; off > 0; off >>= 1) s += __shfl_xor(s, off, 32);
  if (lane == 0) { rmax[gw] = m; rrcp[gw] = 1.f / s; }
}

// ---------- col softmax stats over i (C=1024): one block per (b,j) ----------
__global__ void colstats_kernel(const float* __restrict__ S,
                                float* __restrict__ cmax,
                                float* __restrict__ crcp) {
  __shared__ float red[256];
  int col = blockIdx.x;                 // b*Q + j
  int b = col / Q_, j = col % Q_;
  int tid = threadIdx.x;
  float v[4];
  float m = -1e30f;
#pragma unroll
  for (int s = 0; s < 4; ++s) {
    v[s] = S[(b * C_ + tid + s * 256) * Q_ + j];
    m = fmaxf(m, v[s]);
  }
  red[tid] = m; __syncthreads();
  for (int off = 128; off > 0; off >>= 1) {
    if (tid < off) red[tid] = fmaxf(red[tid], red[tid + off]);
    __syncthreads();
  }
  m = red[0]; __syncthreads();
  float sum = 0.f;
#pragma unroll
  for (int s = 0; s < 4; ++s) sum += __expf(v[s] - m);
  red[tid] = sum; __syncthreads();
  for (int off = 128; off > 0; off >>= 1) {
    if (tid < off) red[tid] += red[tid + off];
    __syncthreads();
  }
  if (tid == 0) { cmax[col] = m; crcp[col] = 1.f / red[0]; }
}

// ------- Tt[b,d,j] = sum_k softmax_col(S)[b,k,j] * ctx[b,d,k]  (M=j, N=d, K=C) -------
// Block: 8 waves share one j-tile; S slab staged via async global->LDS copies.
__global__ void t_gemm_kernel(const float* __restrict__ S,
                              const float* __restrict__ ctx,
                              const float* __restrict__ cmax,
                              const float* __restrict__ crcp,
                              float* __restrict__ Tt) {
  __shared__ float lds_s[32 * 16];        // [k][j] raw S slab, 2 KB

  int tid  = threadIdx.x;
  int w    = tid >> 5;
  int lane = tid & 31;
  int hi = lane >> 4, ln = lane & 15;

  int b  = blockIdx.x >> 3;
  int j0 = (blockIdx.x & 7) * 16;
  int d0 = w * 16;

  float cm = cmax[b * Q_ + j0 + ln];      // A-lane row m = j
  float cr = crcp[b * Q_ + j0 + ln];
  const float* brow = &ctx[(b * D_ + d0 + ln) * C_];  // B col n = d, contiguous in k

  v8f acc = {};
  for (int k0 = 0; k0 < C_; k0 += 32) {
    __syncthreads();
    {   // async-stage 32(k) x 16(j) raw S slab: one b64 per thread
      int flat = tid * 2;
      int k = flat >> 4, jj = flat & 15;
      unsigned loff = (unsigned)(unsigned long long)&lds_s[k * 16 + jj];
      async_copy_b64(loff, &S[(size_t)(b * C_ + k0 + k) * Q_ + j0 + jj]);
    }
    wait_async0();
    __syncthreads();

    v16bf A, Bv;
#pragma unroll
    for (int e = 0; e < 16; ++e) {        // on-the-fly col softmax from LDS
      int k = kslot(e, hi);
      A[e] = (__bf16)(__expf(lds_s[k * 16 + ln] - cm) * cr);
    }
    float4 b0 = *(const float4*)&brow[k0 + hi * 8];
    float4 b1 = *(const float4*)&brow[k0 + hi * 8 + 4];
    float4 b2 = *(const float4*)&brow[k0 + 16 + hi * 8];
    float4 b3 = *(const float4*)&brow[k0 + 20 + hi * 8];
    pack8(Bv, 0, b0, b1); pack8(Bv, 8, b2, b3);
    acc = __builtin_amdgcn_wmma_f32_16x16x32_bf16(false, A, false, Bv,
                                                  (short)0, acc, false, false);
  }
  // store transposed: Tt[b, d, j] so the final pass reads contiguous j
#pragma unroll
  for (int r = 0; r < 8; ++r)
    Tt[(b * D_ + d0 + ln) * Q_ + j0 + r + hi * 8] = acc[r];
}

// ------- Final fused pass: c2q^T = q @ S_row^T ; q2c^T = Tt @ S_row^T -------
// Tile M=d, N=i, K=j (Q=128). All operands contiguous in K -> float4 loads.
__global__ void final_kernel(const float* __restrict__ S,
                             const float* __restrict__ rmax,
                             const float* __restrict__ rrcp,
                             const float* __restrict__ q,
                             const float* __restrict__ Tt,
                             const float* __restrict__ ctx,
                             float* __restrict__ out) {
  int gw   = (blockIdx.x * blockDim.x + threadIdx.x) >> 5;
  int lane = threadIdx.x & 31;
  int hi = lane >> 4, ln = lane & 15;
  const int TC = C_ / 16, TD = D_ / 16;
  int b  = gw / (TD * TC);
  int t  = gw % (TD * TC);
  int d0 = (t / TC) * 16;
  int i0 = (t % TC) * 16;

  float rm = rmax[b * C_ + i0 + ln];      // B-lane col n = i
  float rr = rrcp[b * C_ + i0 + ln];
  const float* srow = &S[(size_t)(b * C_ + i0 + ln) * Q_];
  const float* qrow = &q[(b * D_ + d0 + ln) * Q_];
  const float* trow = &Tt[(b * D_ + d0 + ln) * Q_];

  v8f acc1 = {}, acc2 = {};
#pragma unroll
  for (int k0 = 0; k0 < Q_; k0 += 32) {
    int o0 = k0 + hi * 8, o2 = k0 + 16 + hi * 8;
    float4 s0 = *(const float4*)&srow[o0], s1 = *(const float4*)&srow[o0 + 4];
    float4 s2 = *(const float4*)&srow[o2], s3 = *(const float4*)&srow[o2 + 4];
    v16bf Bv;                             // on-the-fly row softmax
    Bv[0]  = (__bf16)(__expf(s0.x - rm) * rr); Bv[1]  = (__bf16)(__expf(s0.y - rm) * rr);
    Bv[2]  = (__bf16)(__expf(s0.z - rm) * rr); Bv[3]  = (__bf16)(__expf(s0.w - rm) * rr);
    Bv[4]  = (__bf16)(__expf(s1.x - rm) * rr); Bv[5]  = (__bf16)(__expf(s1.y - rm) * rr);
    Bv[6]  = (__bf16)(__expf(s1.z - rm) * rr); Bv[7]  = (__bf16)(__expf(s1.w - rm) * rr);
    Bv[8]  = (__bf16)(__expf(s2.x - rm) * rr); Bv[9]  = (__bf16)(__expf(s2.y - rm) * rr);
    Bv[10] = (__bf16)(__expf(s2.z - rm) * rr); Bv[11] = (__bf16)(__expf(s2.w - rm) * rr);
    Bv[12] = (__bf16)(__expf(s3.x - rm) * rr); Bv[13] = (__bf16)(__expf(s3.y - rm) * rr);
    Bv[14] = (__bf16)(__expf(s3.z - rm) * rr); Bv[15] = (__bf16)(__expf(s3.w - rm) * rr);

    v16bf A1, A2;
    pack8(A1, 0, *(const float4*)&qrow[o0], *(const float4*)&qrow[o0 + 4]);
    pack8(A1, 8, *(const float4*)&qrow[o2], *(const float4*)&qrow[o2 + 4]);
    pack8(A2, 0, *(const float4*)&trow[o0], *(const float4*)&trow[o0 + 4]);
    pack8(A2, 8, *(const float4*)&trow[o2], *(const float4*)&trow[o2 + 4]);

    acc1 = __builtin_amdgcn_wmma_f32_16x16x32_bf16(false, A1, false, Bv,
                                                   (short)0, acc1, false, false);
    acc2 = __builtin_amdgcn_wmma_f32_16x16x32_bf16(false, A2, false, Bv,
                                                   (short)0, acc2, false, false);
  }
  const int BDC = B_ * D_ * C_;
#pragma unroll
  for (int r = 0; r < 8; ++r) {
    int d = d0 + r + hi * 8;
    int base = (b * D_ + d) * C_ + i0 + ln;   // lanes 0-15 contiguous over i
    float cv = ctx[base];
    out[0 * BDC + base] = cv;
    out[1 * BDC + base] = acc1[r];
    out[2 * BDC + base] = cv * acc1[r];
    out[3 * BDC + base] = cv * acc2[r];
  }
}

extern "C" void kernel_launch(void* const* d_in, const int* in_sizes, int n_in,
                              void* d_out, int out_size, void* d_ws, size_t ws_size,
                              hipStream_t stream) {
  const float* ctx  = (const float*)d_in[0];   // [B,D,C]
  const float* q    = (const float*)d_in[1];   // [B,D,Q]
  const float* w_c  = (const float*)d_in[2];   // [D]
  const float* w_q  = (const float*)d_in[3];   // [D]
  const float* w_cq = (const float*)d_in[4];   // [D]
  const float* bias = (const float*)d_in[5];   // [1]
  float* out = (float*)d_out;                  // [4,B,D,C]

  float* ws   = (float*)d_ws;
  float* S    = ws;                         // B*C*Q  = 4,194,304 floats
  float* Tt   = S    + B_ * C_ * Q_;        // B*D*Q  =   524,288
  float* pc   = Tt   + B_ * D_ * Q_;        // B*C
  float* pq   = pc   + B_ * C_;             // B*Q
  float* rmax = pq   + B_ * Q_;             // B*C
  float* rrcp = rmax + B_ * C_;             // B*C
  float* cmax = rrcp + B_ * C_;             // B*Q
  float* crcp = cmax + B_ * Q_;             // B*Q

  const int TPB = 256;                      // 8 wave32 per block

  partc_kernel<<<(B_ * C_ + TPB - 1) / TPB, TPB, 0, stream>>>(ctx, w_c, pc);
  partq_kernel<<<(B_ * Q_ + TPB - 1) / TPB, TPB, 0, stream>>>(q, w_q, pq);

  // S GEMM: B * (C/128) * (Q/16) = 2048 blocks of 8 waves
  s_gemm_kernel<<<B_ * (C_ / 128) * (Q_ / 16), TPB, 0, stream>>>(
      ctx, q, w_cq, pc, pq, bias, S);

  rowstats_kernel<<<(B_ * C_) / 8, TPB, 0, stream>>>(S, rmax, rrcp);
  colstats_kernel<<<B_ * Q_, TPB, 0, stream>>>(S, cmax, crcp);

  // T GEMM: B * (Q/16) = 256 blocks, 8 d-waves each
  t_gemm_kernel<<<B_ * (Q_ / 16), TPB, 0, stream>>>(S, ctx, cmax, crcp, Tt);

  // Final fused pass: 16384 wave-tiles
  final_kernel<<<(B_ * (D_ / 16) * (C_ / 16)) / 8, TPB, 0, stream>>>(
      S, rmax, rrcp, q, Tt, ctx, out);

  (void)in_sizes; (void)n_in; (void)out_size; (void)ws_size;
}
